// MultiHeadAttentionLayer_61598420959242
// MI455X (gfx1250) — matrix operand order repaired and checked by reading, not compiled, not run on hardware
//
#include <hip/hip_runtime.h>

#define N_NODES 50000
#define N_EDGES 500000
#define IN_CH   128
#define OUT_CH  16
#define HEADS   8
#define D_OUT   128  // OUT_CH*HEADS

typedef __bf16 bf16_t;
typedef __attribute__((ext_vector_type(16))) __bf16 v16bf;
typedef __attribute__((ext_vector_type(8)))  float  v8f;

// ---------------------------------------------------------------------------
// Stage a 128x128 f32 weight matrix into LDS as bf16, pre-swizzled into
// B-fragment order for v_wmma_f32_16x16x32_bf16.
// Fragment f = kc*8 + nt (kc = K-chunk 0..3, nt = N-tile 0..7).
// B (32x16) per-lane layout: lane L holds column n = nt*16 + (L&15),
// rows k = kc*32 + 8*(L>>4) + (i&7) + 16*(i>>3) for i=0..15 (32B per lane).
// ---------------------------------------------------------------------------
__device__ __forceinline__ void stage_weight_lds(bf16_t* lds, const float* __restrict__ W) {
  for (int d = threadIdx.x; d < 128 * 128; d += blockDim.x) {
    int f    = d >> 9;          // fragment id (512 bf16 each)
    int lane = (d >> 4) & 31;
    int i    = d & 15;
    int kc   = f >> 3;
    int nt   = f & 7;
    int k    = kc * 32 + ((lane >> 4) << 3) + (i & 7) + ((i >> 3) << 4);
    int n    = nt * 16 + (lane & 15);
    lds[d] = (bf16_t)W[k * 128 + n];
  }
}

// ---------------------------------------------------------------------------
// Build a 16x32 bf16 A-fragment from one f32 row-major tile.
// Per ISA 7.12.2: lane L (L<16) holds row L&15, K in [8h,8h+8) and
// [16+8h,16+8h+8) where h = L>>4.
// ---------------------------------------------------------------------------
__device__ __forceinline__ v16bf load_a_frag(const float* __restrict__ rowbase, int kc, int lane) {
  const int khalf = lane >> 4;
  const float4* p0 = reinterpret_cast<const float4*>(rowbase + kc * 32 + 8 * khalf);
  const float4* p1 = reinterpret_cast<const float4*>(rowbase + kc * 32 + 16 + 8 * khalf);
  float4 a0 = p0[0], a1 = p0[1];
  float4 b0 = p1[0], b1 = p1[1];
  v16bf f;
  f[0]  = (bf16_t)a0.x; f[1]  = (bf16_t)a0.y; f[2]  = (bf16_t)a0.z; f[3]  = (bf16_t)a0.w;
  f[4]  = (bf16_t)a1.x; f[5]  = (bf16_t)a1.y; f[6]  = (bf16_t)a1.z; f[7]  = (bf16_t)a1.w;
  f[8]  = (bf16_t)b0.x; f[9]  = (bf16_t)b0.y; f[10] = (bf16_t)b0.z; f[11] = (bf16_t)b0.w;
  f[12] = (bf16_t)b1.x; f[13] = (bf16_t)b1.y; f[14] = (bf16_t)b1.z; f[15] = (bf16_t)b1.w;
  return f;
}

// Order-independent (deterministic) float atomic max via sign-split trick.
__device__ __forceinline__ void atomicMaxFloat(float* addr, float v) {
  if (v >= 0.0f) atomicMax(reinterpret_cast<int*>(addr), __float_as_int(v));
  else           atomicMin(reinterpret_cast<unsigned int*>(addr), __float_as_uint(v));
}

// ---------------------------------------------------------------------------
// Kernel 0: initialize output / segment buffers.
// ---------------------------------------------------------------------------
__global__ void init_kernel(float* __restrict__ h_out, float* __restrict__ segmax,
                            float* __restrict__ denom) {
  const int tid = blockIdx.x * blockDim.x + threadIdx.x;
  const int stride = gridDim.x * blockDim.x;
  const float ninf = -__builtin_inff();
  for (int i = tid; i < N_NODES * D_OUT; i += stride) h_out[i] = 0.0f;
  for (int i = tid; i < N_NODES * HEADS; i += stride) { segmax[i] = ninf; denom[i] = 0.0f; }
}

// ---------------------------------------------------------------------------
// Kernel 1: node projection  out = X @ W + bias   (one wave per 16-row tile)
// nt loop is NOT unrolled: keeps B-fragment ds_loads inside the loop so LICM
// cannot hoist 32 fragments and spill them to scratch.
// ---------------------------------------------------------------------------
__global__ void proj_kernel(const float* __restrict__ X, const float* __restrict__ W,
                            const float* __restrict__ bias, float* __restrict__ out,
                            int n_tiles) {
  __shared__ bf16_t lds_w[128 * 128];
  stage_weight_lds(lds_w, W);
  __syncthreads();

  const int lane    = threadIdx.x & 31;
  const int wid     = blockIdx.x * (blockDim.x >> 5) + (threadIdx.x >> 5);
  const int wstride = gridDim.x * (blockDim.x >> 5);
  const int lrow    = lane & 15;
  const int lhalf   = lane >> 4;

  for (int tile = wid; tile < n_tiles; tile += wstride) {
    const float* rowbase = X + (size_t)(tile * 16 + lrow) * 128;
    v16bf a0 = load_a_frag(rowbase, 0, lane);
    v16bf a1 = load_a_frag(rowbase, 1, lane);
    v16bf a2 = load_a_frag(rowbase, 2, lane);
    v16bf a3 = load_a_frag(rowbase, 3, lane);
    if (tile + wstride < n_tiles)
      __builtin_prefetch(X + (size_t)((tile + wstride) * 16 + lrow) * 128, 0, 3);

    // base pointer for this lane's output rows: row = tile*16 + 8*lhalf + r
    float* obase = out + (size_t)(tile * 16 + (lhalf << 3)) * 128 + lrow;

#pragma unroll 1
    for (int nt = 0; nt < 8; ++nt) {
      v8f c = {};
#pragma unroll
      for (int kc = 0; kc < 4; ++kc) {
        v16bf b = *reinterpret_cast<const v16bf*>(lds_w + (((kc << 3) + nt) << 9) + (lane << 4));
        const v16bf a = (kc == 0) ? a0 : (kc == 1) ? a1 : (kc == 2) ? a2 : a3;
        c = __builtin_amdgcn_wmma_f32_16x16x32_bf16(false, a, false, b, (short)0, c, false, false);
      }
      const float bv = bias[nt * 16 + lrow];
      float* p = obase + nt * 16;   // column offset within the 128-wide row
#pragma unroll
      for (int r = 0; r < 8; ++r)
        p[r * 128] = c[r] + bv;     // constant byte offsets, single address pair
    }
  }
}

// ---------------------------------------------------------------------------
// Kernel 2: fused edge kernel.
//  E_e tile = e @ We + be via WMMA (never spilled to HBM), then
//  p = K[src]*Q[dst]*E_e*0.25 -> e_out;  per-head score = 16-lane reduction;
//  atomic segment-max into segmax[src,head].  head == nt (OUT_CH==16).
// ---------------------------------------------------------------------------
__global__ void edge_kernel(const float* __restrict__ E, const float* __restrict__ We,
                            const float* __restrict__ be,
                            const int* __restrict__ srcI, const int* __restrict__ dstI,
                            const float* __restrict__ Qh, const float* __restrict__ Kh,
                            float* __restrict__ e_out, float* __restrict__ score,
                            float* __restrict__ segmax, int n_tiles) {
  __shared__ bf16_t lds_w[128 * 128];
  stage_weight_lds(lds_w, We);
  __syncthreads();

  const int lane    = threadIdx.x & 31;
  const int wid     = blockIdx.x * (blockDim.x >> 5) + (threadIdx.x >> 5);
  const int wstride = gridDim.x * (blockDim.x >> 5);
  const int lrow    = lane & 15;
  const int lhalf   = lane >> 4;
  const int mbase   = lhalf << 3;

  for (int tile = wid; tile < n_tiles; tile += wstride) {
    const float* rowbase = E + (size_t)(tile * 16 + lrow) * 128;
    v16bf a0 = load_a_frag(rowbase, 0, lane);
    v16bf a1 = load_a_frag(rowbase, 1, lane);
    v16bf a2 = load_a_frag(rowbase, 2, lane);
    v16bf a3 = load_a_frag(rowbase, 3, lane);
    if (tile + wstride < n_tiles)
      __builtin_prefetch(E + (size_t)((tile + wstride) * 16 + lrow) * 128, 0, 3);

    // Preload the 8 edge indices this lane handles (rows mbase..mbase+7),
    // as int row offsets into the [*,128] f32 arrays.
    const int ebase = tile * 16 + mbase;
    int soff[8], doff[8], sidx[8];
#pragma unroll
    for (int r = 0; r < 8; ++r) {
      sidx[r] = srcI[ebase + r];
      soff[r] = sidx[r] * 128;
      doff[r] = dstI[ebase + r] * 128;
    }
    float* eobase = e_out + (size_t)ebase * 128 + lrow;

#pragma unroll 1
    for (int nt = 0; nt < 8; ++nt) {   // nt == head
      v8f c = {};
#pragma unroll
      for (int kc = 0; kc < 4; ++kc) {
        v16bf b = *reinterpret_cast<const v16bf*>(lds_w + (((kc << 3) + nt) << 9) + (lane << 4));
        const v16bf a = (kc == 0) ? a0 : (kc == 1) ? a1 : (kc == 2) ? a2 : a3;
        c = __builtin_amdgcn_wmma_f32_16x16x32_bf16(false, a, false, b, (short)0, c, false, false);
      }
      const int   col = nt * 16 + lrow;
      const float bv  = be[col];
      float* ep = eobase + nt * 16;
#pragma unroll
      for (int r = 0; r < 8; ++r) {
        const float ee = c[r] + bv;
        const float p  = Kh[soff[r] + col] * Qh[doff[r] + col] * ee * 0.25f;
        ep[r * 128] = p;
        // sum over the 16 channels of this head: butterfly across the 16-lane half
        float acc = p;
        acc += __shfl_xor(acc, 1, 32);
        acc += __shfl_xor(acc, 2, 32);
        acc += __shfl_xor(acc, 4, 32);
        acc += __shfl_xor(acc, 8, 32);
        if (lrow == 0) {
          score[(size_t)(ebase + r) * HEADS + nt] = acc;
          atomicMaxFloat(&segmax[sidx[r] * HEADS + nt], acc);
        }
      }
    }
  }
}

// ---------------------------------------------------------------------------
// Kernel 3: ex = exp(score - segmax[src]); denom[src] += ex (atomic).
// ---------------------------------------------------------------------------
__global__ void softmax_kernel(const int* __restrict__ srcI, const float* __restrict__ segmax,
                               float* __restrict__ score, float* __restrict__ denom) {
  const int tid = blockIdx.x * blockDim.x + threadIdx.x;
  if (tid >= N_EDGES * HEADS) return;
  const int e = tid >> 3;
  const int h = tid & 7;
  const int s = srcI[e];
  const float ex = __expf(score[tid] - segmax[s * HEADS + h]);
  score[tid] = ex;
  atomicAdd(&denom[s * HEADS + h], ex);
}

// ---------------------------------------------------------------------------
// Kernel 4: h_out[src] += (ex/denom[src]) * V[dst]  (f32 global atomics; the
// 25.6MB target and 1.6MB denom live in the 192MB L2).
// ---------------------------------------------------------------------------
__global__ void aggregate_kernel(const int* __restrict__ srcI, const int* __restrict__ dstI,
                                 const float* __restrict__ Vh, const float* __restrict__ score,
                                 const float* __restrict__ denom, float* __restrict__ h_out) {
  const long long tid = (long long)blockIdx.x * blockDim.x + threadIdx.x;
  if (tid >= (long long)N_EDGES * D_OUT) return;
  const int e  = (int)(tid >> 7);
  const int ch = (int)(tid & 127);
  const int h  = ch >> 4;
  const int s  = srcI[e];
  const int d  = dstI[e];
  const float w = score[e * HEADS + h] / denom[s * HEADS + h];
  atomicAdd(&h_out[s * 128 + ch], w * Vh[d * 128 + ch]);
}

// ---------------------------------------------------------------------------
extern "C" void kernel_launch(void* const* d_in, const int* in_sizes, int n_in,
                              void* d_out, int out_size, void* d_ws, size_t ws_size,
                              hipStream_t stream) {
  const float* x  = (const float*)d_in[0];
  const float* e  = (const float*)d_in[1];
  const int*   ei = (const int*)d_in[2];
  const float* Wq = (const float*)d_in[3];
  const float* bq = (const float*)d_in[4];
  const float* Wk = (const float*)d_in[5];
  const float* bk = (const float*)d_in[6];
  const float* Wv = (const float*)d_in[7];
  const float* bv = (const float*)d_in[8];
  const float* We = (const float*)d_in[9];
  const float* be = (const float*)d_in[10];

  float* out   = (float*)d_out;
  float* h_out = out;                                   // [N, 128]
  float* e_out = out + (size_t)N_NODES * D_OUT;         // [E, 128]

  // Workspace layout (f32): Qh,Kh,Vh [N,128] | score [E,8] | segmax,denom [N,8]
  float* Qh     = (float*)d_ws;
  float* Kh     = Qh + (size_t)N_NODES * D_OUT;
  float* Vh     = Kh + (size_t)N_NODES * D_OUT;
  float* score  = Vh + (size_t)N_NODES * D_OUT;
  float* segmax = score + (size_t)N_EDGES * HEADS;
  float* denom  = segmax + (size_t)N_NODES * HEADS;

  const int* srcI = ei;
  const int* dstI = ei + N_EDGES;

  init_kernel<<<1024, 256, 0, stream>>>(h_out, segmax, denom);

  const int node_tiles = N_NODES / 16;   // 3125 (exact)
  const int edge_tiles = N_EDGES / 16;   // 31250 (exact)
  const int pb = (node_tiles + 7) / 8;   // 8 waves/block, one tile per wave
  proj_kernel<<<pb, 256, 0, stream>>>(x, Wq, bq, Qh, node_tiles);
  proj_kernel<<<pb, 256, 0, stream>>>(x, Wk, bk, Kh, node_tiles);
  proj_kernel<<<pb, 256, 0, stream>>>(x, Wv, bv, Vh, node_tiles);

  const int eb = (edge_tiles + 7) / 8;
  edge_kernel<<<eb, 256, 0, stream>>>(e, We, be, srcI, dstI, Qh, Kh,
                                      e_out, score, segmax, edge_tiles);

  softmax_kernel<<<(N_EDGES * HEADS + 255) / 256, 256, 0, stream>>>(srcI, segmax, score, denom);

  const long long total = (long long)N_EDGES * D_OUT;
  aggregate_kernel<<<(int)((total + 255) / 256), 256, 0, stream>>>(srcI, dstI, Vh, score,
                                                                   denom, h_out);
}
